// MCRNN_5428838662835
// MI455X (gfx1250) — compile-verified, matrix operand order checked by reading, not run on hardware
//
#include <hip/hip_runtime.h>
#include <cstdint>

typedef __attribute__((ext_vector_type(16))) _Float16 v16h;
typedef __attribute__((ext_vector_type(8)))  _Float16 v8h;
typedef __attribute__((ext_vector_type(8)))  float    v8f;

#define BATCH 256
#define HID   256
#define TSEQ  32
#define ODIM  96
#define EPS_BN 1e-5f

// k index inside a 16x32 f16 WMMA A/B fragment for element e (0..15) of lane L.
// Contiguous in two 8-element runs: k = [off..off+7] and [16+off..16+off+7],
// off = (lane&16)>>1.
__device__ __forceinline__ int kmap(int e, int lane) {
    return ((e & 8) << 1) | ((lane & 16) >> 1) | (e & 7);
}
__device__ __forceinline__ float sigf(float x) { return 1.f / (1.f + __expf(-x)); }
__device__ __forceinline__ float leakyf(float x) { return x > 0.f ? x : 0.1f * x; }

// Build an A-fragment from a (LDS or global) f16 row: two 16B contiguous loads.
__device__ __forceinline__ v16h frag_from_row(const _Float16* row, int k0, int off) {
    v8h lo = *(const v8h*)(row + k0 + off);
    v8h hi = *(const v8h*)(row + k0 + 16 + off);
    v16h a;
#pragma unroll
    for (int i = 0; i < 8; ++i) { a[i] = lo[i]; a[i + 8] = hi[i]; }
    return a;
}

// ---------------------------------------------------------------------------
// Pre-pass: convert row-major f32 weights (S x N x K) into f16 fragments laid
// out as [s][ntile][kstep][lane][16], zero-padding K up to KP (mult of 32).
// Consumer loads one v16h (32B contiguous) per lane per fragment.
// ---------------------------------------------------------------------------
__global__ __launch_bounds__(256) void swizzle_w_kernel(
    const float* __restrict__ w, _Float16* __restrict__ out,
    int N, int K, int KP, int total) {
    int tid = blockIdx.x * 256 + threadIdx.x;   // over S*N*KP output elements
    if (tid >= total) return;
    int e = tid & 15;
    int r = tid >> 4;
    int lane = r & 31; r >>= 5;
    int ksteps = KP >> 5;
    int kt = r % ksteps; r /= ksteps;
    int ntiles = N >> 4;
    int ntile = r % ntiles;
    int s = r / ntiles;
    int n = ntile * 16 + (lane & 15);
    int k = kt * 32 + kmap(e, lane);
    float val = (k < K) ? w[((size_t)s * N + n) * K + k] : 0.f;
    out[tid] = (_Float16)val;
}

// ---------------------------------------------------------------------------
// Layer 1: Cin=1 direct conv (K=9, too small for WMMA) fused leaky+BN+pool(2,2)
// ---------------------------------------------------------------------------
__global__ __launch_bounds__(256) void conv1_kernel(
    const float* __restrict__ x, const float* __restrict__ w,
    const float* __restrict__ bias, const float* __restrict__ g,
    const float* __restrict__ be, const float* __restrict__ m,
    const float* __restrict__ v, float* __restrict__ out) {
    int lin = blockIdx.x * blockDim.x + threadIdx.x;         // (b,co,hp,wp)
    int wp = lin & 127, t = lin >> 7;
    int hp = t & 15; t >>= 4;
    int co = t & 15; t >>= 4;
    int b = t;
    float scale = g[co] * rsqrtf(v[co] + EPS_BN);
    float mb = m[co], bb = be[co], bi = bias[co];
    const float* xb = x + (size_t)b * 32 * 256;
    const float* wc = w + co * 9;
    float best = -INFINITY;
    for (int hh = 0; hh < 2; ++hh)
        for (int ww = 0; ww < 2; ++ww) {
            int h = hp * 2 + hh, wcol = wp * 2 + ww;
            float acc = bi;
            for (int kh = 0; kh < 3; ++kh) {
                int iy = h + kh - 1;
                if ((unsigned)iy >= 32u) continue;
                for (int kw = 0; kw < 3; ++kw) {
                    int ix = wcol + kw - 1;
                    if ((unsigned)ix >= 256u) continue;
                    acc += xb[iy * 256 + ix] * wc[kh * 3 + kw];
                }
            }
            float yv = leakyf(acc);
            yv = (yv - mb) * scale + bb;
            best = fmaxf(best, yv);
        }
    out[lin] = best;
}

// ---------------------------------------------------------------------------
// Implicit-GEMM conv3x3(SAME) + leaky + BN + maxpool(2,PW), one wave per block.
// A = pre-swizzled f16 weights (one v16h load/frag), B = gathered input patches.
// ---------------------------------------------------------------------------
template <int Cin, int Cout, int Hin, int Win, int PW>
__global__ __launch_bounds__(32) void conv_wmma_kernel(
    const float* __restrict__ X, const _Float16* __restrict__ Wsw,
    const float* __restrict__ bias, const float* __restrict__ g,
    const float* __restrict__ be, const float* __restrict__ m,
    const float* __restrict__ v, float* __restrict__ Out) {
    constexpr int K  = Cin * 9;
    constexpr int KP = ((K + 31) / 32) * 32;
    constexpr int KSTEPS = KP / 32;
    constexpr int Hout = Hin / 2;
    constexpr int Wout = Win / PW;
    constexpr int Wtiles = Win / 16;
    int lane = threadIdx.x;
    int b   = blockIdx.x;
    int cot = blockIdx.y;                 // channel tile
    int co0 = cot * 16;
    int hp  = blockIdx.z / Wtiles;
    int wt  = blockIdx.z % Wtiles;
    int nlane = lane & 15;
    int hi8   = (lane & 16) >> 1;
    const float* xb = X + (size_t)b * Cin * Hin * Win;
    const _Float16* wfrag = Wsw + ((size_t)cot * KSTEPS * 32 + lane) * 16;
    v8f acc0 = {}; v8f acc1 = {};
    for (int ks = 0; ks < KSTEPS; ++ks) {
        int k0 = ks * 32;
        v16h a = *(const v16h*)(wfrag + (size_t)ks * 32 * 16);
        __builtin_prefetch(wfrag + (size_t)(ks + 1) * 32 * 16, 0, 1);
        v16h b0, b1;
#pragma unroll
        for (int e = 0; e < 16; ++e) {
            int k = k0 + kmap(e, lane);
            _Float16 p0 = (_Float16)0.f, p1 = (_Float16)0.f;
            if (k < K) {
                int ci = k / 9, r = k - ci * 9;
                int kh = r / 3, kw = r - kh * 3;
                int ix = wt * 16 + nlane + kw - 1;
                if ((unsigned)ix < (unsigned)Win) {
                    const float* xc = xb + (size_t)ci * Hin * Win + ix;
                    int iy0 = hp * 2 + kh - 1;
                    if ((unsigned)iy0 < (unsigned)Hin) p0 = (_Float16)xc[iy0 * Win];
                    int iy1 = iy0 + 1;
                    if ((unsigned)iy1 < (unsigned)Hin) p1 = (_Float16)xc[iy1 * Win];
                }
            }
            b0[e] = p0; b1[e] = p1;
        }
        acc0 = __builtin_amdgcn_wmma_f32_16x16x32_f16(false, a, false, b0, (short)0, acc0, false, false);
        acc1 = __builtin_amdgcn_wmma_f32_16x16x32_f16(false, a, false, b1, (short)0, acc1, false, false);
    }
#pragma unroll
    for (int r = 0; r < 8; ++r) {
        int co = co0 + r + hi8;
        float sc = g[co] * rsqrtf(v[co] + EPS_BN);
        float y0 = leakyf(acc0[r] + bias[co]); y0 = (y0 - m[co]) * sc + be[co];
        float y1 = leakyf(acc1[r] + bias[co]); y1 = (y1 - m[co]) * sc + be[co];
        float p = fmaxf(y0, y1);                       // pool rows
        if (PW == 2) {
            float o = __shfl_xor(p, 1, 32);            // pool adjacent cols
            p = fmaxf(p, o);
            if ((nlane & 1) == 0) {
                int wout = wt * 8 + (nlane >> 1);
                Out[(((size_t)b * Cout + co) * Hout + hp) * Wout + wout] = p;
            }
        } else {
            int wout = wt * 16 + nlane;
            Out[(((size_t)b * Cout + co) * Hout + hp) * Wout + wout] = p;
        }
    }
}

// ---------------------------------------------------------------------------
// Script classifier: one block per sample, writes probs + argmax script id.
// ---------------------------------------------------------------------------
__global__ __launch_bounds__(256) void script_kernel(
    const float* __restrict__ FEAT,
    const float* c1w, const float* c1b, const float* g1, const float* b1,
    const float* m1, const float* v1,
    const float* c2w, const float* c2b, const float* g2, const float* b2,
    const float* m2, const float* v2,
    const float* f1w, const float* f1b, const float* f2w, const float* f2b,
    float* __restrict__ probs_out, int* __restrict__ sidx) {
    __shared__ float sf[256 * 32];
    __shared__ float s1[128 * 29];
    __shared__ float p1[128 * 7];
    __shared__ float s2[64 * 4];
    __shared__ float p2[64];
    __shared__ float h1[32];
    __shared__ float lg[4];
    int b = blockIdx.x, tid = threadIdx.x;
    for (int i = tid; i < 256 * 32; i += 256) sf[i] = FEAT[(size_t)b * 8192 + i];
    __syncthreads();
    for (int oi = tid; oi < 128 * 29; oi += 256) {       // conv1d 256->128, k=4, VALID
        int co = oi / 29, t = oi - co * 29;
        float acc = c1b[co];
        const float* wr = c1w + (size_t)co * 256 * 4;
        for (int ci = 0; ci < 256; ++ci) {
            const float* fr = sf + ci * 32 + t;
            const float* wc = wr + ci * 4;
            acc += fr[0]*wc[0] + fr[1]*wc[1] + fr[2]*wc[2] + fr[3]*wc[3];
        }
        acc = leakyf(acc);
        acc = (acc - m1[co]) * (g1[co] * rsqrtf(v1[co] + EPS_BN)) + b1[co];
        s1[co * 29 + t] = acc;
    }
    __syncthreads();
    for (int oi = tid; oi < 128 * 7; oi += 256) {        // maxpool1d(4)
        int co = oi / 7, u = oi - co * 7;
        float mx = s1[co * 29 + u * 4];
        for (int q = 1; q < 4; ++q) mx = fmaxf(mx, s1[co * 29 + u * 4 + q]);
        p1[oi] = mx;
    }
    __syncthreads();
    for (int oi = tid; oi < 64 * 4; oi += 256) {         // conv1d 128->64, k=4
        int co = oi >> 2, t = oi & 3;
        float acc = c2b[co];
        const float* wr = c2w + (size_t)co * 128 * 4;
        for (int ci = 0; ci < 128; ++ci) {
            const float* fr = p1 + ci * 7 + t;
            const float* wc = wr + ci * 4;
            acc += fr[0]*wc[0] + fr[1]*wc[1] + fr[2]*wc[2] + fr[3]*wc[3];
        }
        acc = leakyf(acc);
        acc = (acc - m2[co]) * (g2[co] * rsqrtf(v2[co] + EPS_BN)) + b2[co];
        s2[oi] = acc;
    }
    __syncthreads();
    if (tid < 64)                                        // maxpool1d(4) -> 1
        p2[tid] = fmaxf(fmaxf(s2[tid*4], s2[tid*4+1]), fmaxf(s2[tid*4+2], s2[tid*4+3]));
    __syncthreads();
    if (tid < 32) {                                      // fc1 + relu
        float acc = f1b[tid];
        for (int c = 0; c < 64; ++c) acc += p2[c] * f1w[tid * 64 + c];
        h1[tid] = acc > 0.f ? acc : 0.f;
    }
    __syncthreads();
    if (tid < 4) {                                       // fc2
        float acc = f2b[tid];
        for (int j = 0; j < 32; ++j) acc += h1[j] * f2w[tid * 32 + j];
        lg[tid] = acc;
    }
    __syncthreads();
    if (tid == 0) {                                      // softmax + argmax
        float mx = fmaxf(fmaxf(lg[0], lg[1]), fmaxf(lg[2], lg[3]));
        float e0 = __expf(lg[0]-mx), e1 = __expf(lg[1]-mx);
        float e2 = __expf(lg[2]-mx), e3 = __expf(lg[3]-mx);
        float inv = 1.f / (e0 + e1 + e2 + e3);
        probs_out[b*4+0] = e0*inv; probs_out[b*4+1] = e1*inv;
        probs_out[b*4+2] = e2*inv; probs_out[b*4+3] = e3*inv;
        int best = 0; float bv = lg[0];
        if (lg[1] > bv) { bv = lg[1]; best = 1; }
        if (lg[2] > bv) { bv = lg[2]; best = 2; }
        if (lg[3] > bv) { bv = lg[3]; best = 3; }
        sidx[b] = best;
    }
}

// Group samples by script id so LSTM GEMMs are dense per-script tiles.
__global__ void compact_kernel(const int* __restrict__ sidx, int* order, int* cnt, int* offs) {
    if (threadIdx.x == 0 && blockIdx.x == 0) {
        int pos = 0;
        for (int s = 0; s < 4; ++s) {
            offs[s] = pos;
            int c = 0;
            for (int b = 0; b < BATCH; ++b)
                if (sidx[b] == s) { order[pos++] = b; ++c; }
            cnt[s] = c;
        }
    }
}

// ---------------------------------------------------------------------------
// LSTM layer: block = (script s, 16-sample tile). Per timestep two WMMA GEMMs
// (x_t and h) 16x1024x256 accumulated into one v8f; weights are pre-swizzled
// f16 fragments (one 32B load per lane per frag); h/c LDS-resident for 32 steps.
// ---------------------------------------------------------------------------
template <int LAYER>
__global__ __launch_bounds__(256) void lstm_kernel(
    const float* __restrict__ FEAT, const _Float16* __restrict__ Yin,
    const _Float16* __restrict__ wih_sw, const _Float16* __restrict__ whh_sw,
    const float* __restrict__ bih_all, const float* __restrict__ bhh_all,
    const int* __restrict__ order, const int* __restrict__ cnt,
    const int* __restrict__ offs,
    float* __restrict__ HS, float* __restrict__ CS,
    _Float16* __restrict__ Yout) {
    __shared__ _Float16 sh_x[16 * HID];
    __shared__ _Float16 sh_h[16 * HID];
    __shared__ _Float16 sh_G[16 * 4 * HID];
    __shared__ _Float16 sh_c[16 * HID];
    __shared__ int sh_b[16];
    int s  = blockIdx.x;
    int mt = blockIdx.y;
    int vr = cnt[s] - mt * 16;
    if (vr <= 0) return;
    if (vr > 16) vr = 16;
    int tid = threadIdx.x, lane = tid & 31, wv = tid >> 5;
    int nlane = lane & 15, hi8 = (lane & 16) >> 1;
    if (tid < 16) sh_b[tid] = (tid < vr) ? order[offs[s] + mt * 16 + tid] : -1;
    const _Float16* wihs = wih_sw + (size_t)s * 4 * HID * HID;  // 64 ntiles x 8 ksteps
    const _Float16* whhs = whh_sw + (size_t)s * 4 * HID * HID;
    const float* bih = bih_all + s * 4 * HID;
    const float* bhh = bhh_all + s * 4 * HID;
    for (int it = tid; it < 16 * HID; it += 256) {       // init h, c
        int mm = it >> 8, j = it & 255;
        float h0 = 0.f, c0 = 0.f;
        if (LAYER == 1 && mm < vr) {
            int bb = order[offs[s] + mt * 16 + mm];
            h0 = HS[bb * HID + j]; c0 = CS[bb * HID + j];
        }
        sh_h[it] = (_Float16)h0; sh_c[it] = (_Float16)c0;
    }
    __syncthreads();
    for (int t = 0; t < TSEQ; ++t) {
        for (int it = tid; it < 16 * HID; it += 256) {   // gather x_t
            int mm = it >> 8, j = it & 255;
            float xv = 0.f;
            int bb = sh_b[mm];
            if (bb >= 0) {
                if (LAYER == 0) xv = FEAT[((size_t)bb * HID + j) * TSEQ + t];
                else            xv = (float)Yin[((size_t)bb * TSEQ + t) * HID + j];
            }
            sh_x[it] = (_Float16)xv;
        }
        __syncthreads();
        const _Float16* xrow = sh_x + nlane * HID;
        const _Float16* hrow = sh_h + nlane * HID;
        for (int nt = 0; nt < 8; ++nt) {                 // wave wv: gate cols
            int ntile = wv * 8 + nt;
            const _Float16* wfi = wihs + ((size_t)ntile * 8 * 32 + lane) * 16;
            const _Float16* wfh = whhs + ((size_t)ntile * 8 * 32 + lane) * 16;
            __builtin_prefetch(wfi, 0, 1);
            v8f acc = {};
#pragma unroll
            for (int ks = 0; ks < 8; ++ks) {             // x_t @ wih^T
                v16h a  = frag_from_row(xrow, ks * 32, hi8);
                v16h bm = *(const v16h*)(wfi + (size_t)ks * 32 * 16);
                acc = __builtin_amdgcn_wmma_f32_16x16x32_f16(false, a, false, bm, (short)0, acc, false, false);
            }
#pragma unroll
            for (int ks = 0; ks < 8; ++ks) {             // h @ whh^T
                v16h a  = frag_from_row(hrow, ks * 32, hi8);
                v16h bm = *(const v16h*)(wfh + (size_t)ks * 32 * 16);
                acc = __builtin_amdgcn_wmma_f32_16x16x32_f16(false, a, false, bm, (short)0, acc, false, false);
            }
#pragma unroll
            for (int r = 0; r < 8; ++r)
                sh_G[(r + hi8) * (4 * HID) + ntile * 16 + nlane] = (_Float16)acc[r];
        }
        __syncthreads();
        for (int it = tid; it < 16 * HID; it += 256) {   // gates i,f,g,o
            int mm = it >> 8, j = it & 255;
            const _Float16* Gm = sh_G + mm * 4 * HID;
            float gi = (float)Gm[j]           + bih[j]           + bhh[j];
            float gf = (float)Gm[HID + j]     + bih[HID + j]     + bhh[HID + j];
            float gg = (float)Gm[2*HID + j]   + bih[2*HID + j]   + bhh[2*HID + j];
            float go = (float)Gm[3*HID + j]   + bih[3*HID + j]   + bhh[3*HID + j];
            float c = sigf(gf) * (float)sh_c[it] + sigf(gi) * tanhf(gg);
            float h = sigf(go) * tanhf(c);
            sh_c[it] = (_Float16)c;
            sh_h[it] = (_Float16)h;
            int bb = sh_b[mm];
            if (bb >= 0) {
                Yout[((size_t)bb * TSEQ + t) * HID + j] = (_Float16)h;
                if (LAYER == 0 && t == TSEQ - 1) { HS[bb * HID + j] = h; CS[bb * HID + j] = c; }
            }
        }
        __syncthreads();
    }
}

// ---------------------------------------------------------------------------
// Output head: per-sample logits = y2 (32x256) @ fc_w[idx]^T + b, then softmax.
// ---------------------------------------------------------------------------
__global__ __launch_bounds__(64) void fc_head_kernel(
    const _Float16* __restrict__ Y2, const _Float16* __restrict__ fcw_sw,
    const float* __restrict__ fcb_all, const int* __restrict__ sidx,
    float* __restrict__ out) {
    __shared__ float sh_l[TSEQ * ODIM];
    int b = blockIdx.x;
    int tid = threadIdx.x, lane = tid & 31, mt = tid >> 5;
    int s = sidx[b];
    const _Float16* fcs = fcw_sw + (size_t)s * ODIM * HID;   // 6 ntiles x 8 ksteps
    const float* fcb = fcb_all + s * ODIM;
    int nlane = lane & 15, hi8 = (lane & 16) >> 1;
    const _Float16* yrow = Y2 + ((size_t)b * TSEQ + mt * 16 + nlane) * HID;
    for (int nt = 0; nt < 6; ++nt) {
        const _Float16* wf = fcs + ((size_t)nt * 8 * 32 + lane) * 16;
        v8f acc = {};
#pragma unroll
        for (int ks = 0; ks < 8; ++ks) {
            v16h a  = frag_from_row(yrow, ks * 32, hi8);
            v16h bm = *(const v16h*)(wf + (size_t)ks * 32 * 16);
            acc = __builtin_amdgcn_wmma_f32_16x16x32_f16(false, a, false, bm, (short)0, acc, false, false);
        }
#pragma unroll
        for (int r = 0; r < 8; ++r)
            sh_l[(mt * 16 + r + hi8) * ODIM + nt * 16 + nlane] = acc[r];
    }
    __syncthreads();
    if (tid < TSEQ) {
        int t = tid;
        float mx = -INFINITY;
        for (int vv = 0; vv < ODIM; ++vv) mx = fmaxf(mx, sh_l[t * ODIM + vv] + fcb[vv]);
        float sum = 0.f;
        for (int vv = 0; vv < ODIM; ++vv) sum += __expf(sh_l[t * ODIM + vv] + fcb[vv] - mx);
        float inv = 1.f / sum;
        for (int vv = 0; vv < ODIM; ++vv)
            out[((size_t)b * TSEQ + t) * ODIM + vv] = __expf(sh_l[t * ODIM + vv] + fcb[vv] - mx) * inv;
    }
}

// ---------------------------------------------------------------------------
extern "C" void kernel_launch(void* const* d_in, const int* in_sizes, int n_in,
                              void* d_out, int out_size, void* d_ws, size_t ws_size,
                              hipStream_t stream) {
    (void)in_sizes; (void)n_in; (void)out_size; (void)ws_size;
    const float* x = (const float*)d_in[0];
    const float* few[5]; const float* feb[5]; const float* feg[5];
    const float* febe[5]; const float* fem[5]; const float* fev[5];
    for (int i = 0; i < 5; ++i) {
        int base = 1 + 6 * i;
        few[i]  = (const float*)d_in[base + 0];
        feb[i]  = (const float*)d_in[base + 1];
        feg[i]  = (const float*)d_in[base + 2];
        febe[i] = (const float*)d_in[base + 3];
        fem[i]  = (const float*)d_in[base + 4];
        fev[i]  = (const float*)d_in[base + 5];
    }
    int sb = 31;   // sc_params
    const float* c1w = (const float*)d_in[sb+0];  const float* c1b = (const float*)d_in[sb+1];
    const float* g1  = (const float*)d_in[sb+2];  const float* b1  = (const float*)d_in[sb+3];
    const float* m1  = (const float*)d_in[sb+4];  const float* v1  = (const float*)d_in[sb+5];
    const float* c2w = (const float*)d_in[sb+6];  const float* c2b = (const float*)d_in[sb+7];
    const float* g2  = (const float*)d_in[sb+8];  const float* b2  = (const float*)d_in[sb+9];
    const float* m2  = (const float*)d_in[sb+10]; const float* v2  = (const float*)d_in[sb+11];
    const float* f1w = (const float*)d_in[sb+12]; const float* f1b = (const float*)d_in[sb+13];
    const float* f2w = (const float*)d_in[sb+14]; const float* f2b = (const float*)d_in[sb+15];
    int hb = 47;   // head_params
    const float* l1_wih = (const float*)d_in[hb+0]; const float* l1_whh = (const float*)d_in[hb+1];
    const float* l1_bih = (const float*)d_in[hb+2]; const float* l1_bhh = (const float*)d_in[hb+3];
    const float* l2_wih = (const float*)d_in[hb+4]; const float* l2_whh = (const float*)d_in[hb+5];
    const float* l2_bih = (const float*)d_in[hb+6]; const float* l2_bhh = (const float*)d_in[hb+7];
    const float* fc_w   = (const float*)d_in[hb+8]; const float* fc_b   = (const float*)d_in[hb+9];

    // workspace layout
    char* ws = (char*)d_ws;
    size_t off = 0;
    auto take = [&](size_t bytes) -> char* {
        char* p = ws + off;
        off += (bytes + 255) & ~(size_t)255;
        return p;
    };
    float* A0   = (float*)take(sizeof(float) * (size_t)BATCH * 16 * 16 * 128);
    float* A1   = (float*)take(sizeof(float) * (size_t)BATCH * 32 * 8 * 128);
    float* A2   = (float*)take(sizeof(float) * (size_t)BATCH * 64 * 4 * 64);
    float* A3   = (float*)take(sizeof(float) * (size_t)BATCH * 128 * 2 * 32);
    float* FEAT = (float*)take(sizeof(float) * (size_t)BATCH * 256 * 32);
    int*   SIDX = (int*)take(sizeof(int) * BATCH);
    int*   ORD  = (int*)take(sizeof(int) * BATCH);
    int*   CNT  = (int*)take(sizeof(int) * 4);
    int*   OFFS = (int*)take(sizeof(int) * 4);
    float* HS   = (float*)take(sizeof(float) * (size_t)BATCH * HID);
    float* CS   = (float*)take(sizeof(float) * (size_t)BATCH * HID);
    _Float16* Y1 = (_Float16*)take(sizeof(_Float16) * (size_t)BATCH * TSEQ * HID);
    _Float16* Y2 = (_Float16*)take(sizeof(_Float16) * (size_t)BATCH * TSEQ * HID);
    // swizzled f16 weight buffers (output totals = S*N*KP elements)
    _Float16* W2s = (_Float16*)take(sizeof(_Float16) * 32 * 160);
    _Float16* W3s = (_Float16*)take(sizeof(_Float16) * 64 * 288);
    _Float16* W4s = (_Float16*)take(sizeof(_Float16) * 128 * 576);
    _Float16* W5s = (_Float16*)take(sizeof(_Float16) * 256 * 1152);
    _Float16* L1IH = (_Float16*)take(sizeof(_Float16) * 4 * 1024 * 256);
    _Float16* L1HH = (_Float16*)take(sizeof(_Float16) * 4 * 1024 * 256);
    _Float16* L2IH = (_Float16*)take(sizeof(_Float16) * 4 * 1024 * 256);
    _Float16* L2HH = (_Float16*)take(sizeof(_Float16) * 4 * 1024 * 256);
    _Float16* FCS  = (_Float16*)take(sizeof(_Float16) * 4 * 96 * 256);

    float* out_main  = (float*)d_out;                               // (B,32,96)
    float* out_probs = (float*)d_out + (size_t)BATCH * TSEQ * ODIM; // (B,4)

    auto swz = [&](const float* w, _Float16* o, int N, int K, int KP, int S) {
        int total = S * N * KP;
        swizzle_w_kernel<<<(total + 255) / 256, 256, 0, stream>>>(w, o, N, K, KP, total);
    };
    swz(few[1], W2s, 32, 144, 160, 1);
    swz(few[2], W3s, 64, 288, 288, 1);
    swz(few[3], W4s, 128, 576, 576, 1);
    swz(few[4], W5s, 256, 1152, 1152, 1);
    swz(l1_wih, L1IH, 1024, 256, 256, 4);
    swz(l1_whh, L1HH, 1024, 256, 256, 4);
    swz(l2_wih, L2IH, 1024, 256, 256, 4);
    swz(l2_whh, L2HH, 1024, 256, 256, 4);
    swz(fc_w,   FCS,  96,  256, 256, 4);

    // feature extractor
    conv1_kernel<<<(BATCH * 16 * 16 * 128) / 256, 256, 0, stream>>>(
        x, few[0], feb[0], feg[0], febe[0], fem[0], fev[0], A0);
    conv_wmma_kernel<16, 32, 16, 128, 1><<<dim3(BATCH, 2, 8 * 8), 32, 0, stream>>>(
        A0, W2s, feb[1], feg[1], febe[1], fem[1], fev[1], A1);
    conv_wmma_kernel<32, 64, 8, 128, 2><<<dim3(BATCH, 4, 4 * 8), 32, 0, stream>>>(
        A1, W3s, feb[2], feg[2], febe[2], fem[2], fev[2], A2);
    conv_wmma_kernel<64, 128, 4, 64, 2><<<dim3(BATCH, 8, 2 * 4), 32, 0, stream>>>(
        A2, W4s, feb[3], feg[3], febe[3], fem[3], fev[3], A3);
    conv_wmma_kernel<128, 256, 2, 32, 1><<<dim3(BATCH, 16, 1 * 2), 32, 0, stream>>>(
        A3, W5s, feb[4], feg[4], febe[4], fem[4], fev[4], FEAT);

    // script classifier + per-script compaction
    script_kernel<<<BATCH, 256, 0, stream>>>(FEAT,
        c1w, c1b, g1, b1, m1, v1, c2w, c2b, g2, b2, m2, v2,
        f1w, f1b, f2w, f2b, out_probs, SIDX);
    compact_kernel<<<1, 32, 0, stream>>>(SIDX, ORD, CNT, OFFS);

    // 2-layer LSTM with script-selected weights (layer2 seeded with layer1 hT,cT)
    lstm_kernel<0><<<dim3(4, 16), 256, 0, stream>>>(
        FEAT, (const _Float16*)nullptr, L1IH, L1HH, l1_bih, l1_bhh,
        ORD, CNT, OFFS, HS, CS, Y1);
    lstm_kernel<1><<<dim3(4, 16), 256, 0, stream>>>(
        FEAT, Y1, L2IH, L2HH, l2_bih, l2_bhh,
        ORD, CNT, OFFS, HS, CS, Y2);

    // output projection + softmax
    fc_head_kernel<<<BATCH, 64, 0, stream>>>(Y2, FCS, fc_b, SIDX, out_main);
}
